// NNConvNetwork_64244120814372
// MI455X (gfx1250) — compile-verified, matrix operand order checked by reading, not run on hardware
//
#include <hip/hip_runtime.h>

// Problem constants (from reference)
#define NND   16000
#define NE    64000
#define HD    90
#define NODEF 16
#define NGR   512

typedef __attribute__((ext_vector_type(16))) __bf16 v16bf;
typedef __attribute__((ext_vector_type(8)))  float  v8f;

__device__ __forceinline__ unsigned short f2bf(float f) {
  unsigned int u = __float_as_uint(f);
  u += 0x7fffu + ((u >> 16) & 1u);           // round-to-nearest-even
  return (unsigned short)(u >> 16);
}
__device__ __forceinline__ float bf2f(unsigned short s) {
  return __uint_as_float(((unsigned int)s) << 16);
}
// WMMA 16-bit A/B element->K mapping (ISA 7.12.2): v16 element j, lane half
__device__ __forceinline__ int kmap(int j, int half) {
  return (j < 8) ? (j + 8 * half) : (j + 8 + 8 * half);
}

// ---------------- concat x|pos -> xin [N,16] ----------------
__global__ void concat_kernel(const float* __restrict__ x, const float* __restrict__ pos,
                              float* __restrict__ xin, int N) {
  int i = blockIdx.x * blockDim.x + threadIdx.x;
  if (i < N * 16) {
    int n = i >> 4, c = i & 15;
    xin[i] = (c < 13) ? x[n * 13 + c] : pos[n * 3 + (c - 13)];
  }
}

// ---------------- edge MLP: h = relu(ea @ w1 + b1), [E,90] ----------------
__global__ void edge_mlp_kernel(const float* __restrict__ ea, const float* __restrict__ w1,
                                const float* __restrict__ b1, float* __restrict__ h, int E) {
  int e = blockIdx.x;
  int j = threadIdx.x;                     // blockDim = 96
  __shared__ float sea[8];
  if (j < 8) sea[j] = ea[(size_t)e * 8 + j];
  __syncthreads();
  if (j < HD) {
    float acc = b1[j];
#pragma unroll
    for (int i = 0; i < 8; ++i) acc += sea[i] * w1[i * HD + j];
    h[(size_t)e * HD + j] = acc > 0.f ? acc : 0.f;
  }
}

// ---------------- pack A (X[M,K] f32 -> WMMA-A bf16 fragments) ----------------
// fragment id = mt*Kt + kt ; per fragment: 32 lanes x 16 bf16, contiguous (1 KB)
__global__ void pack_a_kernel(const float* __restrict__ X, unsigned short* __restrict__ Ap,
                              int M, int K, int Kt) {
  int frag = blockIdx.x;
  int kt = frag % Kt, mt = frag / Kt;
  int lane = threadIdx.x & 31;
  int half = lane >> 4;
  int m = mt * 16 + (lane & 15);
  size_t off = ((size_t)frag * 32 + lane) * 16;
#pragma unroll
  for (int j = 0; j < 16; ++j) {
    int k = kt * 32 + kmap(j, half);
    float f = (m < M && k < K) ? X[(size_t)m * K + k] : 0.f;
    Ap[off + j] = f2bf(f);
  }
}

// ---------------- pack B (w2 permute + bias row -> WMMA-B bf16 fragments) ------
// B matrix = W' [inC x Nout], Nout=(H+1)*O, W'[i, k*O+o] = w2[k, i*O+o], row H = b2
// fragment id = kt*Nt + nt
__global__ void pack_b_kernel(const float* __restrict__ w2, const float* __restrict__ b2,
                              unsigned short* __restrict__ Bp,
                              int inC, int O, int Nt, int Nout) {
  int frag = blockIdx.x;
  int nt = frag % Nt, kt = frag / Nt;
  int lane = threadIdx.x & 31;
  int half = lane >> 4;
  int c = nt * 16 + (lane & 15);
  int kcol = c / O, o = c - kcol * O;
  size_t off = ((size_t)frag * 32 + lane) * 16;
#pragma unroll
  for (int j = 0; j < 16; ++j) {
    int i = kt * 32 + kmap(j, half);
    float f = 0.f;
    if (c < Nout && i < inC)
      f = (kcol < HD) ? w2[(size_t)kcol * inC * O + (size_t)i * O + o]
                      : b2[(size_t)i * O + o];
    Bp[off + j] = f2bf(f);
  }
}

// ---------------- GEMM: V[M, Nout](bf16) = A(bf16) x B(bf16), f32 accum --------
// KT templated (1 or 3): straight-line fragment loads at constant strides + WMMA chain
template <int KT>
__global__ void gemm_v_kernel(const unsigned short* __restrict__ Ap,
                              const unsigned short* __restrict__ Bp,
                              unsigned short* __restrict__ V,
                              int Mt, int Nt, int Nout) {
  int wave = blockIdx.x * (blockDim.x >> 5) + (threadIdx.x >> 5);
  int lane = threadIdx.x & 31;
  if (wave >= Mt * Nt) return;             // uniform per wave: EXEC stays all-ones
  int mt = wave / Nt, nt = wave - mt * Nt;
  const v16bf* ap = (const v16bf*)(Ap + ((size_t)mt * KT * 32 + lane) * 16);
  const v16bf* bp = (const v16bf*)(Bp + ((size_t)nt * 32 + lane) * 16);
  const size_t bstride = (size_t)Nt * 32;  // fragments (v16bf units = 32 B)
  v8f acc = {};
#pragma unroll
  for (int kt = 0; kt < KT; ++kt) {
    v16bf a = ap[kt * 32];                 // +1 KB per K-step
    v16bf b = bp[kt * bstride];
    acc = __builtin_amdgcn_wmma_f32_16x16x32_bf16(
        /*neg_a=*/false, a, /*neg_b=*/false, b,
        /*c_mod=*/(short)0, acc, /*reuse_a=*/false, /*reuse_b=*/false);
  }
  int col = nt * 16 + (lane & 15);
  int half = lane >> 4;
  if (col < Nout) {
#pragma unroll
    for (int r = 0; r < 8; ++r) {
      size_t row = (size_t)(mt * 16 + r + 8 * half);
      V[row * (size_t)Nout + col] = f2bf(acc[r]);
    }
  }
}

// ---------------- agg init: agg[n,o] = X[n,:] @ root + bias ----------------
__global__ void init_agg_kernel(const float* __restrict__ X, const float* __restrict__ root,
                                const float* __restrict__ bias, float* __restrict__ agg,
                                int K, int O) {
  int n = blockIdx.x, o = threadIdx.x;
  extern __shared__ float sx[];
  for (int i = threadIdx.x; i < K; i += blockDim.x) sx[i] = X[(size_t)n * K + i];
  __syncthreads();
  if (o < O) {
    float acc = bias[o];
    for (int i = 0; i < K; ++i) acc += sx[i] * root[(size_t)i * O + o];
    agg[(size_t)n * O + o] = acc;
  }
}

// ---------------- edge message + scatter-add ----------------
// msg[e,o] = V[src, H*O + o] + sum_k h[e,k]*V[src, k*O + o]; agg[dst,o] += msg
__global__ void edge_msg_kernel(const int* __restrict__ ei, const float* __restrict__ h,
                                const unsigned short* __restrict__ V, float* __restrict__ agg,
                                int E, int O, int Nout) {
  int e = blockIdx.x;
  int o = threadIdx.x;
  __shared__ float shh[HD];
  for (int k = threadIdx.x; k < HD; k += blockDim.x) shh[k] = h[(size_t)e * HD + k];
  __syncthreads();
  int src = ei[e];
  int dst = ei[E + e];
  if (o < O) {
    size_t base = (size_t)src * Nout + o;
    float acc = bf2f(V[base + (size_t)HD * O]);      // absorbed b2 term
    for (int k = 0; k < HD; ++k) acc += shh[k] * bf2f(V[base + (size_t)k * O]);
    atomicAdd(&agg[(size_t)dst * O + o], acc);
  }
}

// ---------------- misc elementwise ----------------
__global__ void relu_kernel(float* __restrict__ p, size_t n) {
  size_t i = (size_t)blockIdx.x * blockDim.x + threadIdx.x;
  if (i < n) p[i] = p[i] > 0.f ? p[i] : 0.f;
}
__global__ void zero_kernel(float* __restrict__ p, size_t n) {
  size_t i = (size_t)blockIdx.x * blockDim.x + threadIdx.x;
  if (i < n) p[i] = 0.f;
}
__global__ void pool_kernel(const float* __restrict__ h3, const int* __restrict__ batch,
                            float* __restrict__ g, int N, int C) {
  int i = blockIdx.x * blockDim.x + threadIdx.x;
  if (i < N * C) {
    int n = i / C, c = i - n * C;
    atomicAdd(&g[(size_t)batch[n] * C + c], h3[i]);
  }
}

// ---------------- head: out[b] = relu(g@fc1 + b) @ out_w + out_b ----------------
__global__ void head_kernel(const float* __restrict__ g, const float* __restrict__ fc1w,
                            const float* __restrict__ fc1b, const float* __restrict__ outw,
                            const float* __restrict__ outb, float* __restrict__ out) {
  int b = blockIdx.x, t = threadIdx.x;     // blockDim = 128
  __shared__ float sg[45];
  __shared__ float red[128];
  if (t < 45) sg[t] = g[b * 45 + t];
  __syncthreads();
  float v = 0.f;
  if (t < HD) {
    float acc = fc1b[t];
#pragma unroll
    for (int j = 0; j < 45; ++j) acc += sg[j] * fc1w[j * HD + t];
    acc = acc > 0.f ? acc : 0.f;
    v = acc * outw[t];
  }
  red[t] = v;
  __syncthreads();
  for (int s = 64; s > 0; s >>= 1) {
    if (t < s) red[t] += red[t + s];
    __syncthreads();
  }
  if (t == 0) out[b] = red[0] + outb[0];
}

extern "C" void kernel_launch(void* const* d_in, const int* in_sizes, int n_in,
                              void* d_out, int out_size, void* d_ws, size_t ws_size,
                              hipStream_t stream) {
  const float* x    = (const float*)d_in[0];
  const float* pos  = (const float*)d_in[1];
  const float* ea   = (const float*)d_in[2];
  const int*   ei   = (const int*)d_in[3];
  const int*   batch= (const int*)d_in[4];
  const float* w1[3]   = {(const float*)d_in[5],  (const float*)d_in[11], (const float*)d_in[17]};
  const float* b1[3]   = {(const float*)d_in[6],  (const float*)d_in[12], (const float*)d_in[18]};
  const float* w2[3]   = {(const float*)d_in[7],  (const float*)d_in[13], (const float*)d_in[19]};
  const float* b2[3]   = {(const float*)d_in[8],  (const float*)d_in[14], (const float*)d_in[20]};
  const float* root[3] = {(const float*)d_in[9],  (const float*)d_in[15], (const float*)d_in[21]};
  const float* bias[3] = {(const float*)d_in[10], (const float*)d_in[16], (const float*)d_in[22]};
  const float* fc1w = (const float*)d_in[23];
  const float* fc1b = (const float*)d_in[24];
  const float* outw = (const float*)d_in[25];
  const float* outb = (const float*)d_in[26];
  float* out = (float*)d_out;

  // workspace bump allocator
  char* ws = (char*)d_ws;
  size_t off = 0;
  auto alloc = [&](size_t bytes) -> void* {
    void* p = ws + off;
    off = (off + bytes + 255) & ~((size_t)255);
    return p;
  };
  const int Mt = NND / 16;                      // 1000
  float* xin = (float*)alloc((size_t)NND * NODEF * 4);
  float* hbuf[3];
  for (int c = 0; c < 3; ++c) hbuf[c] = (float*)alloc((size_t)NE * HD * 4);
  unsigned short* Ap = (unsigned short*)alloc((size_t)Mt * 3 * 512 * 2);
  unsigned short* Bp = (unsigned short*)alloc((size_t)3 * 512 * 512 * 2);
  unsigned short* V  = (unsigned short*)alloc((size_t)NND * ((HD + 1) * HD) * 2);
  float* bufP = (float*)alloc((size_t)NND * HD * 4);
  float* bufQ = (float*)alloc((size_t)NND * HD * 4);
  float* g = (float*)alloc((size_t)NGR * 45 * 4);

  // stage 0: concat + edge MLPs
  concat_kernel<<<(NND * 16 + 255) / 256, 256, 0, stream>>>(x, pos, xin, NND);
  for (int c = 0; c < 3; ++c)
    edge_mlp_kernel<<<NE, 96, 0, stream>>>(ea, w1[c], b1[c], hbuf[c], NE);

  // per-layer configs
  const float* Xl[3]  = {xin, bufP, bufQ};
  float*       Agg[3] = {bufP, bufQ, bufP};
  const int Kl[3]   = {NODEF, HD, HD};
  const int Ol[3]   = {HD, HD, 45};
  for (int c = 0; c < 3; ++c) {
    int K = Kl[c], O = Ol[c];
    int Kt = (K + 31) / 32;                     // 1, 3, 3
    int Nout = (HD + 1) * O;
    int Nt = (Nout + 15) / 16;                  // 512, 512, 256
    int eb = (O == 45) ? 64 : 96;               // edge/agg block size (wave multiples)
    pack_b_kernel<<<Kt * Nt, 32, 0, stream>>>(w2[c], b2[c], Bp, K, O, Nt, Nout);
    pack_a_kernel<<<Mt * Kt, 32, 0, stream>>>(Xl[c], Ap, NND, K, Kt);
    int gblocks = (Mt * Nt + 7) / 8;
    if (Kt == 1)
      gemm_v_kernel<1><<<gblocks, 256, 0, stream>>>(Ap, Bp, V, Mt, Nt, Nout);
    else
      gemm_v_kernel<3><<<gblocks, 256, 0, stream>>>(Ap, Bp, V, Mt, Nt, Nout);
    init_agg_kernel<<<NND, eb, K * sizeof(float), stream>>>(Xl[c], root[c], bias[c], Agg[c], K, O);
    edge_msg_kernel<<<NE, eb, 0, stream>>>(ei, hbuf[c], V, Agg[c], NE, O, Nout);
    relu_kernel<<<((size_t)NND * O + 255) / 256, 256, 0, stream>>>(Agg[c], (size_t)NND * O);
  }

  // pooling + head
  zero_kernel<<<(NGR * 45 + 255) / 256, 256, 0, stream>>>(g, (size_t)NGR * 45);
  pool_kernel<<<(NND * 45 + 255) / 256, 256, 0, stream>>>(Agg[2], batch, g, NND, 45);
  head_kernel<<<NGR, 128, 0, stream>>>(g, fc1w, fc1b, outw, outb, out);
}
